// TransformerBlock_47081431498930
// MI455X (gfx1250) — matrix operand order repaired and checked by reading, not compiled, not run on hardware
//
#include <hip/hip_runtime.h>
#include <hip/hip_bf16.h>

// ---------------------------------------------------------------------------
// Transformer block for MI455X (gfx1250, wave32, WMMA).
// All GEMMs use v_wmma_f32_16x16x32_bf16 (bf16 in, fp32 accumulate).
// Residuals / softmax / rmsnorm statistics stay in fp32.
// Double-buffered LDS + register prefetch, one barrier per K-step.
// Flash attention issues next-tile loads BEFORE the barrier so the prefetch
// registers die at the staging store right after it (no spills across the
// softmax pressure peak); the barrier wait hides the load latency.
// Workspace requirement: 128 MB.
// ---------------------------------------------------------------------------

typedef __bf16 bf16;
typedef __attribute__((ext_vector_type(16))) __bf16 v16bf;
typedef __attribute__((ext_vector_type(8)))  float  v8f;

#define D_MODEL 1024
#define N_HEADS 16
#define DKH     64
#define D_FF    4096
#define SEQ     2048
#define BATCH   2
#define ROWS    (BATCH * SEQ)   // 4096 token rows

static __device__ __forceinline__ v8f v8f_zero() {
    v8f z = {0.f, 0.f, 0.f, 0.f, 0.f, 0.f, 0.f, 0.f};
    return z;
}

// wave-local LDS fence: DS ops of one wave are in-order at the LDS; this only
// has to stop compiler reordering and wait out the VGPR->LDS latency.
static __device__ __forceinline__ void lds_fence_wave() {
    asm volatile("s_wait_dscnt 0x0" ::: "memory");
}

// A-fragment K index for VGPR pair v (contiguous k order; B fragments use the same):
//   k = half*16 + v*2   (holds k, k+1)
#define AFRAG_K(v, half) ((half) * 16 + (v) * 2)

// ---------------------------------------------------------------------------
// fp32 -> bf16 conversion (weights)
// ---------------------------------------------------------------------------
__global__ __launch_bounds__(256)
void cvt_f32_to_bf16(const float* __restrict__ in, bf16* __restrict__ out, int n) {
    int i = blockIdx.x * blockDim.x + threadIdx.x;
    if (i < n) out[i] = (bf16)in[i];
}

// ---------------------------------------------------------------------------
// RMSNorm over rows of length d, output bf16 (GEMM operand precision).
// ---------------------------------------------------------------------------
__global__ __launch_bounds__(256)
void rmsnorm_to_bf16(const float* __restrict__ x, const float* __restrict__ g,
                     bf16* __restrict__ out, int d) {
    const int row = blockIdx.x;
    const float* xr = x + (size_t)row * d;
    float s = 0.f;
    for (int i = threadIdx.x; i < d; i += blockDim.x) {
        float v = xr[i];
        s += v * v;
    }
    #pragma unroll
    for (int m = 16; m >= 1; m >>= 1) s += __shfl_xor(s, m, 32);
    __shared__ float red[8];
    const int wid = threadIdx.x >> 5;
    if ((threadIdx.x & 31) == 0) red[wid] = s;
    __syncthreads();
    float tot = 0.f;
    #pragma unroll
    for (int wv = 0; wv < 8; ++wv) tot += red[wv];
    const float inv = rsqrtf(tot / (float)d + 1e-5f);
    bf16* orow = out + (size_t)row * d;
    for (int i = threadIdx.x; i < d; i += blockDim.x)
        orow[i] = (bf16)(xr[i] * inv * g[i]);
}

// ---------------------------------------------------------------------------
// NT GEMM: C[M,N] = A[M,KC] * B[N,KC]^T  (+ optional fp32 residual)
// 128x64 tile / 128 threads (4 waves); each wave owns 32(M) x 64(N):
// per K-step (32): 2 A frags + 4 B frags -> 8 WMMAs.
// Double-buffered LDS, one barrier per K-step, branchless loop body.
// ---------------------------------------------------------------------------
template<int RESID, int KC>
__global__ __launch_bounds__(128, 1)
void gemm_nt(const bf16* __restrict__ A, const bf16* __restrict__ B,
             const float* __restrict__ resid, float* __restrict__ C, int N) {
    constexpr int NK = KC / 32;
    __shared__ __align__(16) bf16 As[2][128 * 32];   // 2 x 8 KB
    __shared__ __align__(16) bf16 Bs[2][64 * 32];    // 2 x 4 KB

    const int t    = threadIdx.x;
    const int lane = t & 31;
    const int w    = t >> 5;          // wave id -> 32-row M strip
    const int half = lane >> 4;
    const int mrow = lane & 15;
    const int m0   = blockIdx.y * 128;
    const int n0   = blockIdx.x * 64;

    // cooperative-load slots: seg = t&3 (16B within 32-elem row), rbase = t>>2
    const int seg   = t & 3;
    const int rbase = t >> 2;         // 0..31
    const bf16* Abase = A + (size_t)(m0 + rbase) * KC + seg * 8;
    const bf16* Bbase = B + (size_t)(n0 + rbase) * KC + seg * 8;
    uint4* sA = (uint4*)(&As[0][rbase * 32]) + seg;  // +512/buf, +128/strip
    uint4* sB = (uint4*)(&Bs[0][rbase * 32]) + seg;  // +256/buf, +128/strip

    uint4 ra[4], rb[2];
    auto load_tile = [&](int ko) {    // ko: element offset along K
        #pragma unroll
        for (int j = 0; j < 4; ++j)
            ra[j] = *(const uint4*)(Abase + ko + j * (32 * KC));
        #pragma unroll
        for (int j = 0; j < 2; ++j)
            rb[j] = *(const uint4*)(Bbase + ko + j * (32 * KC));
    };
    auto store_tile = [&](int p) {
        #pragma unroll
        for (int j = 0; j < 4; ++j) sA[p * 512 + j * 128] = ra[j];
        #pragma unroll
        for (int j = 0; j < 2; ++j) sB[p * 256 + j * 128] = rb[j];
    };

    v8f acc[2][4];
    #pragma unroll
    for (int mi = 0; mi < 2; ++mi)
        #pragma unroll
        for (int nt = 0; nt < 4; ++nt) acc[mi][nt] = v8f_zero();

    auto compute = [&](int p) {
        const bf16* AsP = &As[p][0];
        const bf16* BsP = &Bs[p][0];
        v16bf a0, a1;
        #pragma unroll
        for (int v = 0; v < 8; ++v) {
            int k = AFRAG_K(v, half);
            const bf16* p0 = &AsP[(w * 32 + mrow) * 32 + k];
            const bf16* p1 = p0 + 16 * 32;
            a0[2 * v] = p0[0]; a0[2 * v + 1] = p0[1];
            a1[2 * v] = p1[0]; a1[2 * v + 1] = p1[1];
        }
        #pragma unroll
        for (int nt = 0; nt < 4; ++nt) {
            v16bf b;
            #pragma unroll
            for (int v = 0; v < 8; ++v) {
                int k = half * 16 + 2 * v;
                const bf16* p = &BsP[(nt * 16 + mrow) * 32 + k];
                b[2 * v] = p[0]; b[2 * v + 1] = p[1];
            }
            acc[0][nt] = __builtin_amdgcn_wmma_f32_16x16x32_bf16(
                false, a0, false, b, (short)0, acc[0][nt], false, false);
            acc[1][nt] = __builtin_amdgcn_wmma_f32_16x16x32_bf16(
                false, a1, false, b, (short)0, acc[1][nt], false, false);
        }
    };

    load_tile(0);                     // tile 0 -> regs
    store_tile(0);                    // regs -> buf 0
    int p = 0;
    #pragma unroll 1
    for (int kt = 0; kt < NK; ++kt) {
        __syncthreads();              // buf[p] ready; prior buf[1-p] reads done
        // branchless: last iteration reloads tile 0 and dead-stores buf[p^1]
        load_tile((kt + 1 < NK) ? (kt + 1) * 32 : 0);
        compute(p);                   // single BB: stores interleave w/ WMMAs
        store_tile(p ^ 1);
        p ^= 1;
    }

    // C layout: M = v + 8*half, N = mrow
    #pragma unroll
    for (int mi = 0; mi < 2; ++mi) {
        #pragma unroll
        for (int nt = 0; nt < 4; ++nt) {
            #pragma unroll
            for (int v = 0; v < 8; ++v) {
                int r  = m0 + w * 32 + mi * 16 + v + 8 * half;
                int cc = n0 + nt * 16 + mrow;
                size_t idx = (size_t)r * N + cc;
                float val = acc[mi][nt][v];
                if (RESID) val += resid[idx];
                C[idx] = val;
            }
        }
    }
}

// ---------------------------------------------------------------------------
// Fused SwiGLU GEMM: Out[M,N] = bf16( silu(A*W1^T) * (A*W3^T) )
// 64x64 tile / 128 threads; per wave 16x64 for both branches: 8 WMMA/K-step.
// ---------------------------------------------------------------------------
template<int KC>
__global__ __launch_bounds__(128, 1)
void gemm_nt_swiglu(const bf16* __restrict__ A, const bf16* __restrict__ B1,
                    const bf16* __restrict__ B3, bf16* __restrict__ Out, int N) {
    constexpr int NK = KC / 32;
    __shared__ __align__(16) bf16 As[2][64 * 32];
    __shared__ __align__(16) bf16 B1s[2][64 * 32];
    __shared__ __align__(16) bf16 B3s[2][64 * 32];

    const int t    = threadIdx.x;
    const int lane = t & 31;
    const int w    = t >> 5;
    const int half = lane >> 4;
    const int mrow = lane & 15;
    const int m0   = blockIdx.y * 64;
    const int n0   = blockIdx.x * 64;

    const int seg   = t & 3;
    const int rbase = t >> 2;
    const bf16* Abase  = A  + (size_t)(m0 + rbase) * KC + seg * 8;
    const bf16* B1base = B1 + (size_t)(n0 + rbase) * KC + seg * 8;
    const bf16* B3base = B3 + (size_t)(n0 + rbase) * KC + seg * 8;
    uint4* sA  = (uint4*)(&As[0][rbase * 32])  + seg;   // +256/buf, +128/strip
    uint4* sB1 = (uint4*)(&B1s[0][rbase * 32]) + seg;
    uint4* sB3 = (uint4*)(&B3s[0][rbase * 32]) + seg;

    uint4 ra[2], rb1[2], rb3[2];
    auto load_tile = [&](int ko) {
        #pragma unroll
        for (int j = 0; j < 2; ++j) {
            ra[j]  = *(const uint4*)(Abase  + ko + j * (32 * KC));
            rb1[j] = *(const uint4*)(B1base + ko + j * (32 * KC));
            rb3[j] = *(const uint4*)(B3base + ko + j * (32 * KC));
        }
    };
    auto store_tile = [&](int p) {
        #pragma unroll
        for (int j = 0; j < 2; ++j) {
            sA[p * 256 + j * 128]  = ra[j];
            sB1[p * 256 + j * 128] = rb1[j];
            sB3[p * 256 + j * 128] = rb3[j];
        }
    };

    v8f au[4], av[4];
    #pragma unroll
    for (int i = 0; i < 4; ++i) { au[i] = v8f_zero(); av[i] = v8f_zero(); }

    auto compute = [&](int p) {
        const bf16* AsP  = &As[p][0];
        const bf16* B1sP = &B1s[p][0];
        const bf16* B3sP = &B3s[p][0];
        v16bf a;
        #pragma unroll
        for (int v = 0; v < 8; ++v) {
            int k = AFRAG_K(v, half);
            const bf16* pp = &AsP[(w * 16 + mrow) * 32 + k];
            a[2 * v] = pp[0]; a[2 * v + 1] = pp[1];
        }
        #pragma unroll
        for (int nt = 0; nt < 4; ++nt) {
            v16bf b1, b3;
            #pragma unroll
            for (int v = 0; v < 8; ++v) {
                int k = half * 16 + 2 * v;
                const bf16* p1 = &B1sP[(nt * 16 + mrow) * 32 + k];
                const bf16* p3 = &B3sP[(nt * 16 + mrow) * 32 + k];
                b1[2 * v] = p1[0]; b1[2 * v + 1] = p1[1];
                b3[2 * v] = p3[0]; b3[2 * v + 1] = p3[1];
            }
            au[nt] = __builtin_amdgcn_wmma_f32_16x16x32_bf16(
                false, a, false, b1, (short)0, au[nt], false, false);
            av[nt] = __builtin_amdgcn_wmma_f32_16x16x32_bf16(
                false, a, false, b3, (short)0, av[nt], false, false);
        }
    };

    load_tile(0);
    store_tile(0);
    int p = 0;
    #pragma unroll 1
    for (int kt = 0; kt < NK; ++kt) {
        __syncthreads();
        load_tile((kt + 1 < NK) ? (kt + 1) * 32 : 0);
        compute(p);
        store_tile(p ^ 1);
        p ^= 1;
    }

    #pragma unroll
    for (int nt = 0; nt < 4; ++nt) {
        #pragma unroll
        for (int v = 0; v < 8; ++v) {
            int r  = m0 + w * 16 + v + 8 * half;
            int cc = n0 + nt * 16 + mrow;
            float u  = au[nt][v];
            float vv = av[nt][v];
            float gate = u / (1.f + __expf(-u));     // silu(u)
            Out[(size_t)r * N + cc] = (bf16)(gate * vv);
        }
    }
}

// ---------------------------------------------------------------------------
// RoPE + head pack: fp32 [row=b*S+s, h*64+j]  ->  bf16 [(b*H+h), s, j]
// ---------------------------------------------------------------------------
__global__ __launch_bounds__(256)
void rope_pack(const float* __restrict__ Qf, const float* __restrict__ Kf,
               const float* __restrict__ Vf, bf16* __restrict__ Qb,
               bf16* __restrict__ Kb, bf16* __restrict__ Vb) {
    int gid = blockIdx.x * blockDim.x + threadIdx.x;   // [row][head][pair]
    int i   = gid & 31;
    int h   = (gid >> 5) & 15;
    int row = gid >> 9;
    if (row >= ROWS) return;
    int s = row & (SEQ - 1);
    int b = row >> 11;

    size_t in_base  = (size_t)row * D_MODEL + h * DKH;
    size_t out_base = (((size_t)(b * N_HEADS + h)) * SEQ + s) * DKH;

    float inv_freq = __powf(10000.f, -(float)i / 32.f);
    float ang = (float)s * inv_freq;
    float sn, cs;
    __sincosf(ang, &sn, &cs);

    float q1 = Qf[in_base + 2 * i], q2 = Qf[in_base + 2 * i + 1];
    float k1 = Kf[in_base + 2 * i], k2 = Kf[in_base + 2 * i + 1];
    Qb[out_base + 2 * i]     = (bf16)(q1 * cs - q2 * sn);
    Qb[out_base + 2 * i + 1] = (bf16)(q1 * sn + q2 * cs);
    Kb[out_base + 2 * i]     = (bf16)(k1 * cs - k2 * sn);
    Kb[out_base + 2 * i + 1] = (bf16)(k1 * sn + k2 * cs);
    Vb[out_base + 2 * i]     = (bf16)Vf[in_base + 2 * i];
    Vb[out_base + 2 * i + 1] = (bf16)Vf[in_base + 2 * i + 1];
}

// ---------------------------------------------------------------------------
// Flash attention (causal).  Grid: (S/64 q-tiles, B*H).  Block: 128 (4 waves).
// Double-buffered 32-key K/V tiles; ONE block barrier per tile.  Next-tile
// global loads are issued BEFORE the barrier and staged to LDS right after
// it, so the prefetch registers never live across the softmax pressure peak.
// P-restage fused into softmax; wave-local dscnt fence (Ps private per wave).
// ---------------------------------------------------------------------------
__global__ __launch_bounds__(128, 1)
void flash_attn(const bf16* __restrict__ Qb, const bf16* __restrict__ Kb,
                const bf16* __restrict__ Vb, bf16* __restrict__ Out) {
    __shared__ __align__(16) bf16 Ks[2][32 * 64];
    __shared__ __align__(16) bf16 Vs[2][32 * 64];
    __shared__ __align__(16) bf16 Ps[4 * 16 * 32];   // per-wave P staging

    const int t    = threadIdx.x;
    const int lane = t & 31;
    const int w    = t >> 5;
    const int half = lane >> 4;
    const int mrow = lane & 15;
    const int bh   = blockIdx.y;            // b*16 + h
    const int b    = bh >> 4;
    const int h    = bh & 15;
    const int qbase = blockIdx.x * 64;
    const int qr    = qbase + w * 16 + mrow;   // A-fragment row for this lane

    const bf16* Qg = Qb + (size_t)bh * SEQ * DKH;
    const bf16* Kg = Kb + (size_t)bh * SEQ * DKH;
    const bf16* Vg = Vb + (size_t)bh * SEQ * DKH;

    // Q A-fragments (dk 0..31 and 32..63), loaded once
    v16bf aq0, aq1;
    {
        const bf16* qp = Qg + (size_t)qr * DKH;
        #pragma unroll
        for (int v = 0; v < 8; ++v) {
            int k = AFRAG_K(v, half);
            aq0[2 * v]     = qp[k];
            aq0[2 * v + 1] = qp[k + 1];
            aq1[2 * v]     = qp[32 + k];
            aq1[2 * v + 1] = qp[32 + k + 1];
        }
    }

    // cooperative K/V tile slots: row = (t>>3)+16*j, seg = t&7
    const int segKV = t & 7;
    const int rKV   = t >> 3;               // 0..15
    const bf16* Kbase = Kg + (size_t)rKV * DKH + segKV * 8;
    const bf16* Vbase = Vg + (size_t)rKV * DKH + segKV * 8;
    uint4* sK = (uint4*)(&Ks[0][rKV * DKH]) + segKV;   // +256/buf, +128/strip
    uint4* sV = (uint4*)(&Vs[0][rKV * DKH]) + segKV;

    uint4 rk[2], rv[2];
    auto load_tile = [&](int ko) {          // ko: element offset (2048/tile)
        #pragma unroll
        for (int j = 0; j < 2; ++j) {
            rk[j] = *(const uint4*)(Kbase + ko + j * 1024);
            rv[j] = *(const uint4*)(Vbase + ko + j * 1024);
        }
    };
    auto store_tile = [&](int p) {
        #pragma unroll
        for (int j = 0; j < 2; ++j) {
            sK[p * 256 + j * 128] = rk[j];
            sV[p * 256 + j * 128] = rv[j];
        }
    };

    v8f acc[4];
    #pragma unroll
    for (int i = 0; i < 4; ++i) acc[i] = v8f_zero();
    float mstat[8], lstat[8];
    #pragma unroll
    for (int v = 0; v < 8; ++v) { mstat[v] = -1e30f; lstat[v] = 0.f; }

    auto compute = [&](int kb, int p) {
        const bf16* KsP = &Ks[p][0];
        const bf16* VsP = &Vs[p][0];

        // S = Q * K_tile^T  (two 16x16 key tiles, two dk K-steps each)
        v8f s0 = v8f_zero(), s1 = v8f_zero();
        #pragma unroll
        for (int nt = 0; nt < 2; ++nt) {
            v16bf b0, b1;
            #pragma unroll
            for (int v = 0; v < 8; ++v) {
                int kd = half * 16 + 2 * v;
                const bf16* pp = &KsP[(nt * 16 + mrow) * DKH + kd];
                b0[2 * v] = pp[0];  b0[2 * v + 1] = pp[1];     // dk 0..31
                b1[2 * v] = pp[32]; b1[2 * v + 1] = pp[33];    // dk 32..63
            }
            v8f* sd = nt ? &s1 : &s0;
            *sd = __builtin_amdgcn_wmma_f32_16x16x32_bf16(
                false, aq0, false, b0, (short)0, *sd, false, false);
            *sd = __builtin_amdgcn_wmma_f32_16x16x32_bf16(
                false, aq1, false, b1, (short)0, *sd, false, false);
        }

        // online softmax with fused P restage; row = v + 8*half,
        // col = kb + nt*16 + mrow.  e0/e1 die inside the iteration.
        #pragma unroll
        for (int v = 0; v < 8; ++v) {
            int rq  = qbase + w * 16 + v + 8 * half;
            float sv0 = ((kb + mrow) <= rq)      ? s0[v] * 0.125f : -1e30f;
            float sv1 = ((kb + 16 + mrow) <= rq) ? s1[v] * 0.125f : -1e30f;
            float mt = fmaxf(sv0, sv1);
            #pragma unroll
            for (int d = 1; d < 16; d <<= 1) mt = fmaxf(mt, __shfl_xor(mt, d, 32));
            float mnew = fmaxf(mstat[v], mt);
            float scl  = __expf(mstat[v] - mnew);
            float e0   = __expf(sv0 - mnew);
            float e1   = __expf(sv1 - mnew);
            float rs   = e0 + e1;
            #pragma unroll
            for (int d = 1; d < 16; d <<= 1) rs += __shfl_xor(rs, d, 32);
            lstat[v] = lstat[v] * scl + rs;
            mstat[v] = mnew;
            bf16* pp = &Ps[w * 512 + (v + 8 * half) * 32 + mrow];
            pp[0]  = (bf16)e0;
            pp[16] = (bf16)e1;
            #pragma unroll
            for (int nt = 0; nt < 4; ++nt) acc[nt][v] = acc[nt][v] * scl;
        }
        lds_fence_wave();                   // wave-local: Ps write -> read

        // P A-fragment (16x32) and acc += P * V_tile
        v16bf pa;
        #pragma unroll
        for (int v = 0; v < 8; ++v) {
            int k = AFRAG_K(v, half);
            const bf16* pp = &Ps[w * 512 + mrow * 32 + k];
            pa[2 * v]     = pp[0];
            pa[2 * v + 1] = pp[1];
        }
        #pragma unroll
        for (int nt = 0; nt < 4; ++nt) {
            v16bf bv;
            #pragma unroll
            for (int v = 0; v < 8; ++v) {
                int kr = half * 16 + 2 * v;          // key row within tile
                bv[2 * v]     = VsP[kr * DKH + nt * 16 + mrow];
                bv[2 * v + 1] = VsP[(kr + 1) * DKH + nt * 16 + mrow];
            }
            acc[nt] = __builtin_amdgcn_wmma_f32_16x16x32_bf16(
                false, pa, false, bv, (short)0, acc[nt], false, false);
        }
    };

    const int nkt = blockIdx.x * 2 + 2;     // covers keys 0 .. qbase+63
    load_tile(0);
    store_tile(0);
    int p = 0;
    #pragma unroll 1
    for (int kt = 0; kt < nkt; ++kt) {
        // issue next-tile loads BEFORE the barrier: their latency overlaps
        // the barrier wait, and the registers die at store_tile just after.
        load_tile((kt + 1 < nkt) ? (kt + 1) * 2048 : 0);
        __syncthreads();                    // buf[p] ready; buf[1-p] reads done
        store_tile(p ^ 1);                  // regs -> other buffer; regs die
        compute(kt * 32, p);
        p ^= 1;
    }

    // epilogue: normalize and scatter to [b, q, h*64+dk] bf16
    #pragma unroll
    for (int v = 0; v < 8; ++v) {
        int rq = qbase + w * 16 + v + 8 * half;
        float inv = 1.f / lstat[v];
        #pragma unroll
        for (int nt = 0; nt < 4; ++nt) {
            Out[((size_t)(b * SEQ + rq)) * D_MODEL + h * DKH + nt * 16 + mrow] =
                (bf16)(acc[nt][v] * inv);
        }
    }
}

// ---------------------------------------------------------------------------
// Launcher.  Workspace layout (128 MB total, with reuse):
//   [0,32M)    bf16 weights: Wq Wk Wv Wo (2M each), W1 W3 W2 (8M each)
//   [32,40M)   hbf   (rmsnorm output; reused for h2)
//   [40,56M)   Qf fp32      | later: gbf (bf16, spans 40..72M)
//   [56,72M)   Kf fp32      |
//   [72,88M)   Vf fp32      | later: attnbf (bf16, 8M)
//   [88,96M)   Qb bf16 [B,H,S,64]
//   [96,104M)  Kb bf16
//   [104,112M) Vb bf16
//   [112,128M) x1 fp32 (post-attention residual)
// ---------------------------------------------------------------------------
extern "C" void kernel_launch(void* const* d_in, const int* in_sizes, int n_in,
                              void* d_out, int out_size, void* d_ws, size_t ws_size,
                              hipStream_t stream) {
    const float* x  = (const float*)d_in[0];
    const float* Wq = (const float*)d_in[1];
    const float* Wk = (const float*)d_in[2];
    const float* Wv = (const float*)d_in[3];
    const float* Wo = (const float*)d_in[4];
    const float* g1 = (const float*)d_in[5];
    const float* g2 = (const float*)d_in[6];
    const float* W1 = (const float*)d_in[7];
    const float* W3 = (const float*)d_in[8];
    const float* W2 = (const float*)d_in[9];
    float* out = (float*)d_out;
    char* ws = (char*)d_ws;
    const size_t MB = 1024u * 1024u;

    bf16* wq_bf  = (bf16*)(ws + 0 * MB);
    bf16* wk_bf  = (bf16*)(ws + 2 * MB);
    bf16* wv_bf  = (bf16*)(ws + 4 * MB);
    bf16* wo_bf  = (bf16*)(ws + 6 * MB);
    bf16* w1_bf  = (bf16*)(ws + 8 * MB);
    bf16* w3_bf  = (bf16*)(ws + 16 * MB);
    bf16* w2_bf  = (bf16*)(ws + 24 * MB);
    bf16* hbf    = (bf16*)(ws + 32 * MB);
    float* Qf    = (float*)(ws + 40 * MB);
    float* Kf    = (float*)(ws + 56 * MB);
    float* Vf    = (float*)(ws + 72 * MB);
    bf16* gbf    = (bf16*)(ws + 40 * MB);     // reuses Qf+Kf after RoPE pack
    bf16* attnbf = (bf16*)(ws + 72 * MB);     // reuses Vf after RoPE pack
    bf16* Qb     = (bf16*)(ws + 88 * MB);
    bf16* Kb     = (bf16*)(ws + 96 * MB);
    bf16* Vb     = (bf16*)(ws + 104 * MB);
    float* x1    = (float*)(ws + 112 * MB);

    // weight conversion to bf16
    cvt_f32_to_bf16<<<4096, 256, 0, stream>>>(Wq, wq_bf, D_MODEL * D_MODEL);
    cvt_f32_to_bf16<<<4096, 256, 0, stream>>>(Wk, wk_bf, D_MODEL * D_MODEL);
    cvt_f32_to_bf16<<<4096, 256, 0, stream>>>(Wv, wv_bf, D_MODEL * D_MODEL);
    cvt_f32_to_bf16<<<4096, 256, 0, stream>>>(Wo, wo_bf, D_MODEL * D_MODEL);
    cvt_f32_to_bf16<<<16384, 256, 0, stream>>>(W1, w1_bf, D_FF * D_MODEL);
    cvt_f32_to_bf16<<<16384, 256, 0, stream>>>(W3, w3_bf, D_FF * D_MODEL);
    cvt_f32_to_bf16<<<16384, 256, 0, stream>>>(W2, w2_bf, D_MODEL * D_FF);

    // h1 = rmsnorm(x, g1)
    rmsnorm_to_bf16<<<ROWS, 256, 0, stream>>>(x, g1, hbf, D_MODEL);

    // Q/K/V projections (fp32 out)
    dim3 g_proj(D_MODEL / 64, ROWS / 128);   // (16, 32)
    gemm_nt<0, D_MODEL><<<g_proj, 128, 0, stream>>>(hbf, wq_bf, nullptr, Qf, D_MODEL);
    gemm_nt<0, D_MODEL><<<g_proj, 128, 0, stream>>>(hbf, wk_bf, nullptr, Kf, D_MODEL);
    gemm_nt<0, D_MODEL><<<g_proj, 128, 0, stream>>>(hbf, wv_bf, nullptr, Vf, D_MODEL);

    // RoPE + pack to [B,H,S,dk] bf16
    rope_pack<<<(ROWS * N_HEADS * 32) / 256, 256, 0, stream>>>(Qf, Kf, Vf, Qb, Kb, Vb);

    // causal flash attention
    flash_attn<<<dim3(SEQ / 64, BATCH * N_HEADS), 128, 0, stream>>>(Qb, Kb, Vb, attnbf);

    // x1 = x + attn * Wo^T
    gemm_nt<1, D_MODEL><<<g_proj, 128, 0, stream>>>(attnbf, wo_bf, x, x1, D_MODEL);

    // h2 = rmsnorm(x1, g2)
    rmsnorm_to_bf16<<<ROWS, 256, 0, stream>>>(x1, g2, hbf, D_MODEL);

    // g = silu(h2*W1^T) * (h2*W3^T)   (bf16 out)
    gemm_nt_swiglu<D_MODEL><<<dim3(D_FF / 64, ROWS / 64), 128, 0, stream>>>(
        hbf, w1_bf, w3_bf, gbf, D_FF);

    // out = x1 + g * W2^T
    gemm_nt<1, D_FF><<<g_proj, 128, 0, stream>>>(gbf, w2_bf, x1, out, D_MODEL);
}
